// AssignAttention_24309514895551
// MI455X (gfx1250) — matrix operand order repaired and checked by reading, not compiled, not run on hardware
//
#include <hip/hip_runtime.h>

typedef __attribute__((ext_vector_type(16))) __bf16 v16bf;
typedef __attribute__((ext_vector_type(8)))  __bf16 v8bf;
typedef __attribute__((ext_vector_type(4)))  __bf16 v4bf;
typedef __attribute__((ext_vector_type(8)))  float  v8f;

// Problem constants
#define BB 16
#define NN 64
#define SS 4096
#define CC 384
#define HH 6
#define DH 64

// ---------------------------------------------------------------------------
// gfx1250 async-copy plumbing (inline asm; ASYNCcnt-tracked LDS staging).
// Falls back to synchronous copies off-device so the host pass still parses.
// ---------------------------------------------------------------------------
#if defined(__gfx1250__)
#define USE_ASYNC_LDS 1
#else
#define USE_ASYNC_LDS 0
#endif

// One wave-wide async 16B/lane copy: global -> LDS, tracked by ASYNCcnt.
__device__ __forceinline__ void async_copy_b128(const void* gsrc, void* ldst) {
#if USE_ASYNC_LDS
  const unsigned lds_addr = (unsigned)(uintptr_t)ldst;  // low 32b == LDS offset
  asm volatile("global_load_async_to_lds_b128 %0, %1, off"
               :
               : "v"(lds_addr), "v"(gsrc)
               : "memory");
#else
  *reinterpret_cast<uint4*>(ldst) = *reinterpret_cast<const uint4*>(gsrc);
#endif
}

template <int N>
__device__ __forceinline__ void wait_async() {
#if USE_ASYNC_LDS
  asm volatile("s_wait_asynccnt %0" ::"i"(N) : "memory");
#endif
}

// Stage a MT x 32 bf16 tile of X (row stride CC) into LDS buffer (row stride 40).
template <int MT>
__device__ __forceinline__ void issue_tile(const __bf16* __restrict__ X, int mBase,
                                           int k0, __bf16* buf, int tid) {
#pragma unroll
  for (int c = tid; c < MT * 4; c += 256) {
    const int row = c >> 2, cc = c & 3;
    const __bf16* src = X + (size_t)(mBase + row) * CC + k0 + cc * 8;
    __bf16* dst = &buf[row * 40 + cc * 8];
    async_copy_b128(src, dst);
  }
}

// A fragment (16x32 bf16): lane L<16 -> M=L, K in {0..15}; L>=16 -> K in {16..31}
__device__ __forceinline__ v16bf load_afrag(const __bf16* buf, int row, int khalf) {
  v8bf lo = *reinterpret_cast<const v8bf*>(buf + row * 40 + khalf * 16);
  v8bf hi = *reinterpret_cast<const v8bf*>(buf + row * 40 + 8 + khalf * 16);
  v16bf a;
#pragma unroll
  for (int e = 0; e < 8; ++e) { a[e] = lo[e]; a[e + 8] = hi[e]; }
  return a;
}

__device__ __forceinline__ v8f wmma_bf16(v16bf a, v16bf b, v8f c) {
  return __builtin_amdgcn_wmma_f32_16x16x32_bf16(false, a, false, b, (short)0, c,
                                                 false, false);
}

// ---------------------------------------------------------------------------
// f32 -> bf16 conversion (vectorized x4)
// ---------------------------------------------------------------------------
__global__ __launch_bounds__(256) void cvt_f32_bf16(const float* __restrict__ x,
                                                    __bf16* __restrict__ y, int n4) {
  int t = blockIdx.x * 256 + threadIdx.x;
  if (t >= n4) return;
  float4 f = reinterpret_cast<const float4*>(x)[t];
  v4bf o;
  o[0] = (__bf16)f.x; o[1] = (__bf16)f.y; o[2] = (__bf16)f.z; o[3] = (__bf16)f.w;
  reinterpret_cast<v4bf*>(y)[t] = o;
}

// ---------------------------------------------------------------------------
// Out[M x 384] = X[M x 384] @ W[384 x 384]^T (optionally + bias), bf16 WMMA.
// Block: 256 threads (8 waves). Tile: 256 rows x 64 cols, double-buffered
// async LDS staging of X; B fragments read straight from W rows (contiguous
// k order, as in the A fragments). Wave w owns rows {w*16.., 128+w*16..} and all
// four 16-col subtiles -> 8 WMMAs per K-step per wave.
// ---------------------------------------------------------------------------
template <bool BIAS, typename OutT>
__global__ __launch_bounds__(256) void gemm_xwt(const __bf16* __restrict__ X,
                                                const __bf16* __restrict__ W,
                                                const float* __restrict__ bias,
                                                OutT* __restrict__ Out) {
  constexpr int KD = CC, MT = 256;
  __shared__ __bf16 xs[2][MT * 40];

  const int tid = threadIdx.x, lane = tid & 31, w = tid >> 5;
  const int mBase = blockIdx.x * MT, colBase = blockIdx.y * 64;
  const int l16 = lane & 15, khalf = lane >> 4;

  v8f acc[2][4] = {};

  issue_tile<MT>(X, mBase, 0, xs[0], tid);

  for (int kb = 0; kb < 12; ++kb) {
    const int k0 = kb * 32;
    const __bf16* cur = xs[kb & 1];
    if (kb < 11) {
      issue_tile<MT>(X, mBase, k0 + 32, xs[(kb + 1) & 1], tid);
      wait_async<4>();  // only the 4 just-issued next-tile loads stay pending
    } else {
      wait_async<0>();
    }
    __syncthreads();

    const v16bf af0 = load_afrag(cur, w * 16 + l16, khalf);
    const v16bf af1 = load_afrag(cur, 128 + w * 16 + l16, khalf);
#pragma unroll
    for (int ct = 0; ct < 4; ++ct) {
      const int brow = colBase + ct * 16 + l16;
      const v16bf bf = *reinterpret_cast<const v16bf*>(
          W + (size_t)brow * KD + k0 + khalf * 16);
      acc[0][ct] = wmma_bf16(af0, bf, acc[0][ct]);
      acc[1][ct] = wmma_bf16(af1, bf, acc[1][ct]);
    }
    __syncthreads();
  }

  // Epilogue: C/D layout lane -> N=l16; VGPR j -> M = j + 8*khalf
#pragma unroll
  for (int mh = 0; mh < 2; ++mh) {
#pragma unroll
    for (int ct = 0; ct < 4; ++ct) {
      const int col = colBase + ct * 16 + l16;
      const float bv = BIAS ? bias[col] : 0.0f;
#pragma unroll
      for (int j = 0; j < 8; ++j) {
        const int m = mBase + mh * 128 + w * 16 + j + 8 * khalf;
        Out[(size_t)m * KD + col] = (OutT)(acc[mh][ct][j] + bv);
      }
    }
  }
}

// ---------------------------------------------------------------------------
// For each (b,h,s): idx = argmax_n <q_n, k_s>. One wave handles 16 s-columns,
// computing all 4 n-tiles via WMMA; accumulator holds, per lane, 32 of the 64
// n-scores for its s. One shfl_xor(16) merges the complementary half.
// Grid: (32 s-chunks of 128, B*H), 256 threads.
// ---------------------------------------------------------------------------
__global__ __launch_bounds__(256) void attn_argmax_kernel(const __bf16* __restrict__ Qp,
                                                          const __bf16* __restrict__ Kp,
                                                          int* __restrict__ idx) {
  const int tid = threadIdx.x, lane = tid & 31, w = tid >> 5;
  const int i = blockIdx.y, b = i / HH, h = i % HH;
  const int l16 = lane & 15, khalf = lane >> 4;
  const int sBase = (blockIdx.x * 8 + w) * 16;

  const size_t krow = (size_t)(b * SS + sBase + l16) * CC + h * DH;
  v8f acc[4] = {};
#pragma unroll
  for (int kb = 0; kb < 2; ++kb) {
    const int k0 = kb * 32;
    const v16bf bf = *reinterpret_cast<const v16bf*>(Kp + krow + k0 + khalf * 16);
#pragma unroll
    for (int nt = 0; nt < 4; ++nt) {
      const size_t qrow = (size_t)(b * NN + nt * 16 + l16) * CC + h * DH;
      v8bf lo = *reinterpret_cast<const v8bf*>(Qp + qrow + k0 + khalf * 16);
      v8bf hi = *reinterpret_cast<const v8bf*>(Qp + qrow + k0 + 8 + khalf * 16);
      v16bf af;
#pragma unroll
      for (int e = 0; e < 8; ++e) { af[e] = lo[e]; af[e + 8] = hi[e]; }
      acc[nt] = wmma_bf16(af, bf, acc[nt]);
    }
  }

  float best = -3.4e38f;
  int bestn = 0;
#pragma unroll
  for (int nt = 0; nt < 4; ++nt)
#pragma unroll
    for (int j = 0; j < 8; ++j) {
      const int n = nt * 16 + j + 8 * khalf;
      const float v = acc[nt][j];
      if (v > best || (v == best && n < bestn)) { best = v; bestn = n; }
    }
  const float ov = __shfl_xor(best, 16, 32);
  const int   on = __shfl_xor(bestn, 16, 32);
  if (ov > best || (ov == best && on < bestn)) { best = ov; bestn = on; }

  if (lane < 16) idx[(size_t)i * SS + sBase + lane] = bestn;
}

// ---------------------------------------------------------------------------
// Scatter-accumulate v_s into group buckets via LDS atomics, flush to global.
// Grid: (4 s-chunks of 1024, B*H), 256 threads. LDS: 64x64 f32 + 64 counts.
// ---------------------------------------------------------------------------
__global__ __launch_bounds__(256) void scatter_kernel(const __bf16* __restrict__ Vp,
                                                      const int* __restrict__ idx,
                                                      float* __restrict__ acc,
                                                      int* __restrict__ cnt) {
  __shared__ float sacc[NN * DH];
  __shared__ int scnt[NN];
  const int tid = threadIdx.x, lane = tid & 31, w = tid >> 5;
  const int i = blockIdx.y, b = i / HH, h = i % HH;

  for (int e = tid; e < NN * DH; e += 256) sacc[e] = 0.0f;
  if (tid < NN) scnt[tid] = 0;
  __syncthreads();

  const int s0 = blockIdx.x * 1024 + w * 128;
  for (int t = 0; t < 128; ++t) {
    const int s = s0 + t;
    const int n = idx[(size_t)i * SS + s];
    const size_t row = (size_t)(b * SS + s) * CC + h * DH;
    const float v0 = (float)Vp[row + lane];
    const float v1 = (float)Vp[row + 32 + lane];
    atomicAdd(&sacc[n * DH + lane], v0);
    atomicAdd(&sacc[n * DH + 32 + lane], v1);
    if (lane == 0) atomicAdd(&scnt[n], 1);
  }
  __syncthreads();

  for (int e = tid; e < NN * DH; e += 256) atomicAdd(&acc[(size_t)i * NN * DH + e], sacc[e]);
  if (tid < NN) atomicAdd(&cnt[i * NN + tid], scnt[tid]);
}

// ---------------------------------------------------------------------------
// O[b,n,h*64+d] = acc[b,h,n,d] / (cnt[b,h,n] + 1), written as bf16.
// ---------------------------------------------------------------------------
__global__ __launch_bounds__(256) void normalize_kernel(const float* __restrict__ acc,
                                                        const int* __restrict__ cnt,
                                                        __bf16* __restrict__ O) {
  const int t = blockIdx.x * 256 + threadIdx.x;
  if (t >= BB * HH * NN * DH) return;
  const int d = t & (DH - 1);
  const int n = (t >> 6) & (NN - 1);
  const int i = t >> 12;  // b*H + h
  const int b = i / HH, h = i % HH;
  const float v = acc[t] / ((float)cnt[i * NN + n] + 1.0f);
  O[(size_t)(b * NN + n) * CC + h * DH + d] = (__bf16)v;
}

// ---------------------------------------------------------------------------
extern "C" void kernel_launch(void* const* d_in, const int* in_sizes, int n_in,
                              void* d_out, int out_size, void* d_ws, size_t ws_size,
                              hipStream_t stream) {
  const float* query = (const float*)d_in[0];  // [16,64,384]
  const float* key   = (const float*)d_in[1];  // [16,4096,384]
  const float* Wq    = (const float*)d_in[2];  // [384,384]
  const float* Wk    = (const float*)d_in[3];
  const float* Wv    = (const float*)d_in[4];
  const float* Wp    = (const float*)d_in[5];
  const float* bp    = (const float*)d_in[6];  // [384]
  (void)in_sizes; (void)n_in; (void)out_size; (void)ws_size;

  // Workspace carve-out (256B aligned)
  char* p = (char*)d_ws;
  auto carve = [&](size_t bytes) -> void* {
    void* r = (void*)p;
    p += (bytes + 255) & ~(size_t)255;
    return r;
  };
  const size_t nKey = (size_t)BB * SS * CC;
  const size_t nQ   = (size_t)BB * NN * CC;
  const size_t nW   = (size_t)CC * CC;

  __bf16* keyB = (__bf16*)carve(nKey * 2);
  __bf16* qB   = (__bf16*)carve(nQ * 2);
  __bf16* WqB  = (__bf16*)carve(nW * 2);
  __bf16* WkB  = (__bf16*)carve(nW * 2);
  __bf16* WvB  = (__bf16*)carve(nW * 2);
  __bf16* WpB  = (__bf16*)carve(nW * 2);
  __bf16* Kp   = (__bf16*)carve(nKey * 2);
  __bf16* Vp   = (__bf16*)carve(nKey * 2);
  __bf16* Qp   = (__bf16*)carve(nQ * 2);
  __bf16* OB   = (__bf16*)carve(nQ * 2);
  int*    idx  = (int*)carve((size_t)BB * HH * SS * 4);
  float*  acc  = (float*)carve((size_t)BB * HH * NN * DH * 4);
  int*    cnt  = (int*)carve((size_t)BB * HH * NN * 4);

  auto cvt = [&](const float* src, __bf16* dst, size_t n) {
    const int n4 = (int)(n / 4);
    cvt_f32_bf16<<<(n4 + 255) / 256, 256, 0, stream>>>(src, dst, n4);
  };
  cvt(key, keyB, nKey);
  cvt(query, qB, nQ);
  cvt(Wq, WqB, nW);
  cvt(Wk, WkB, nW);
  cvt(Wv, WvB, nW);
  cvt(Wp, WpB, nW);

  (void)hipMemsetAsync(acc, 0, (size_t)BB * HH * NN * DH * 4, stream);
  (void)hipMemsetAsync(cnt, 0, (size_t)BB * HH * NN * 4, stream);

  // Projections (bf16 WMMA GEMMs, 256-row macro tiles)
  gemm_xwt<false, __bf16><<<dim3((BB * SS) / 256, CC / 64), 256, 0, stream>>>(keyB, WkB, nullptr, Kp);
  gemm_xwt<false, __bf16><<<dim3((BB * SS) / 256, CC / 64), 256, 0, stream>>>(keyB, WvB, nullptr, Vp);
  gemm_xwt<false, __bf16><<<dim3((BB * NN) / 256, CC / 64), 256, 0, stream>>>(qB, WqB, nullptr, Qp);

  // Hard-assignment argmax over groups
  attn_argmax_kernel<<<dim3(SS / 128, BB * HH), 256, 0, stream>>>(Qp, Kp, idx);

  // Segmented accumulate + counts
  scatter_kernel<<<dim3(SS / 1024, BB * HH), 256, 0, stream>>>(Vp, idx, acc, cnt);

  // Normalize -> bf16 O in [B,N,C] layout
  normalize_kernel<<<(BB * HH * NN * DH + 255) / 256, 256, 0, stream>>>(acc, cnt, OB);

  // Final projection with bias -> f32 output
  gemm_xwt<true, float><<<dim3((BB * NN) / 256, CC / 64), 256, 0, stream>>>(OB, WpB, bp, (float*)d_out);
}
